// RC_STML_39994735460956
// MI455X (gfx1250) — 16-bit and 8-bit matrix operand order repaired and checked by reading, compiled, not run on hardware
//
#include <hip/hip_runtime.h>
#include <hip/hip_bf16.h>
#include <math.h>

#define NROWS 4096
#define NDIM  512

typedef __attribute__((ext_vector_type(16))) __bf16 v16bf;
typedef __attribute__((ext_vector_type(8)))  __bf16 v8bf;
typedef __attribute__((ext_vector_type(8)))  float  v8f;
typedef __attribute__((ext_vector_type(8)))  int    v8i;
typedef __attribute__((ext_vector_type(4)))  int    v4i;
typedef __attribute__((ext_vector_type(4)))  unsigned int v4u;

#if defined(__has_builtin)
#if __has_builtin(__builtin_amdgcn_tensor_load_to_lds)
#define HAVE_TDM 1
#endif
#endif

__device__ __forceinline__ unsigned short f2bf(float x) {
  unsigned int u = __float_as_uint(x);
  if ((u & 0x7f800000u) == 0x7f800000u) return (unsigned short)(u >> 16);
  unsigned int r = u + 0x7fffu + ((u >> 16) & 1u);
  return (unsigned short)(r >> 16);
}

#ifdef HAVE_TDM
__device__ __forceinline__ void tdm_load_tile_u8(const unsigned char* gptr,
                                                 unsigned lds_off,
                                                 unsigned rows, unsigned n) {
  unsigned long long ga = (unsigned long long)(size_t)gptr;
  v4u g0;
  g0[0] = 1u;
  g0[1] = lds_off;
  g0[2] = (unsigned)ga;
  g0[3] = (unsigned)((ga >> 32) & 0x01FFFFFFu)
          | (2u << 30);
  v8i g1;
  g1[0] = 0;
  g1[1] = (int)((n & 0xFFFFu) << 16);
  g1[2] = (int)(((n >> 16) & 0xFFFFu) | ((n & 0xFFFFu) << 16));
  g1[3] = (int)(((n >> 16) & 0xFFFFu) | (64u << 16));
  g1[4] = (int)(rows & 0xFFFFu);
  g1[5] = (int)n;
  g1[6] = 0;
  g1[7] = 0;
  v4i z4 = {0, 0, 0, 0};
#if __clang_major__ >= 23
  v8i z8 = {0, 0, 0, 0, 0, 0, 0, 0};
  __builtin_amdgcn_tensor_load_to_lds(g0, g1, z4, z4, z8, 0);
#else
  __builtin_amdgcn_tensor_load_to_lds(g0, g1, z4, z4, 0);
#endif
}
#else
__device__ __forceinline__ void stage_tile_u8(const unsigned char* __restrict__ g,
                                              unsigned char* __restrict__ lds,
                                              int rows, int n, int tid) {
  const int chunks = rows * 4;
  for (int c = tid; c < chunks; c += 256) {
    int r = c >> 2, o = (c & 3) * 16;
    *(int4*)(lds + r * 64 + o) = *(const int4*)(g + (size_t)r * n + o);
  }
}
#endif

__global__ __launch_bounds__(256)
void k_prep(const float* __restrict__ s, const float* __restrict__ t,
            unsigned short* __restrict__ sbf, unsigned short* __restrict__ tbf,
            float* __restrict__ sq_s, int n, int d) {
  const int i = blockIdx.x;
  const int tid = threadIdx.x;
  __shared__ float red[256];
  const float* srow = s + (size_t)i * d;
  const float* trow = t + (size_t)i * d;
  float ls = 0.f, lt = 0.f;
  for (int j = tid; j < d; j += 256) {
    float a = srow[j]; ls += a * a;
    float b = trow[j]; lt += b * b;
  }
  red[tid] = ls; __syncthreads();
  for (int st = 128; st > 0; st >>= 1) { if (tid < st) red[tid] += red[tid + st]; __syncthreads(); }
  float snorm2 = red[0]; __syncthreads();
  red[tid] = lt; __syncthreads();
  for (int st = 128; st > 0; st >>= 1) { if (tid < st) red[tid] += red[tid + st]; __syncthreads(); }
  float tnorm2 = red[0];
  float tinv = 1.0f / fmaxf(sqrtf(tnorm2), 1e-12f);
  if (tid == 0) sq_s[i] = snorm2;
  for (int j = tid; j < d; j += 256) {
    sbf[(size_t)i * d + j] = f2bf(srow[j]);
    tbf[(size_t)i * d + j] = f2bf(trow[j] * tinv);
  }
}

__global__ __launch_bounds__(256)
void k_gemm_bf16_syrk(const __bf16* __restrict__ A, float* __restrict__ C, int n, int d) {
  const int lane = threadIdx.x & 31;
  const int wave = threadIdx.x >> 5;
  const int h   = lane >> 4;
  const int l16 = lane & 15;
  const int wm = wave & 3, wn = wave >> 2;
  const int m0 = blockIdx.x * 128 + wm * 32;
  const int n0 = blockIdx.y * 64  + wn * 32;

  v8f acc[2][2] = {};
  for (int k = 0; k < d; k += 32) {
    v16bf a[2], b[2];
#pragma unroll
    for (int i = 0; i < 2; ++i) {
      const __bf16* arow = A + (size_t)(m0 + 16 * i + l16) * d + k;
      v8bf lo = *(const v8bf*)(arow + h * 8);
      v8bf hi = *(const v8bf*)(arow + 16 + h * 8);
#pragma unroll
      for (int e = 0; e < 8; ++e) { a[i][e] = lo[e]; a[i][e + 8] = hi[e]; }
      __builtin_prefetch(arow + 32, 0, 1);
    }
#pragma unroll
    for (int j = 0; j < 2; ++j) {
      const __bf16* brow = A + (size_t)(n0 + 16 * j + l16) * d + k;
      v8bf blo = *(const v8bf*)(brow + h * 8);
      v8bf bhi = *(const v8bf*)(brow + 16 + h * 8);
#pragma unroll
      for (int e = 0; e < 8; ++e) { b[j][e] = blo[e]; b[j][e + 8] = bhi[e]; }
    }
#pragma unroll
    for (int i = 0; i < 2; ++i)
#pragma unroll
      for (int j = 0; j < 2; ++j)
        acc[i][j] = __builtin_amdgcn_wmma_f32_16x16x32_bf16(
            false, a[i], false, b[j], (short)0, acc[i][j], false, false);
  }
#pragma unroll
  for (int i = 0; i < 2; ++i)
#pragma unroll
    for (int j = 0; j < 2; ++j) {
      int col = n0 + 16 * j + l16;
#pragma unroll
      for (int r = 0; r < 8; ++r) {
        int row = m0 + 16 * i + h * 8 + r;
        C[(size_t)row * n + col] = acc[i][j][r];
      }
    }
}

__global__ __launch_bounds__(256)
void k_sdist(float* __restrict__ G, const float* __restrict__ sq,
             float* __restrict__ rmean, int n) {
  const int i = blockIdx.x;
  const int tid = threadIdx.x;
  __shared__ float red[256];
  float sqi = sq[i];
  float lsum = 0.f;
  for (int j = tid; j < n; j += 256) {
    size_t e = (size_t)i * n + j;
    float d2 = sqi + sq[j] - 2.0f * G[e];
    float dist = (d2 > 0.0f) ? sqrtf(d2) : 0.0f;
    G[e] = dist;
    lsum += dist;
  }
  red[tid] = lsum; __syncthreads();
  for (int st = 128; st > 0; st >>= 1) { if (tid < st) red[tid] += red[tid + st]; __syncthreads(); }
  if (tid == 0) rmean[i] = red[0] / (float)n;
}

__global__ __launch_bounds__(256)
void k_wp(float* __restrict__ G, unsigned int total) {
  for (unsigned int e = blockIdx.x * 256u + threadIdx.x; e < total; e += gridDim.x * 256u) {
    float d2 = fmaxf(2.0f - 2.0f * G[e], 0.0f);
    G[e] = expf(-d2);
  }
}

__global__ __launch_bounds__(256)
void k_topk(const float* __restrict__ WP, const int* __restrict__ idx,
            int* __restrict__ topk, unsigned char* __restrict__ WNN, int n) {
  const int i = blockIdx.x;
  const int tid = threadIdx.x;
  __shared__ float sval[256];
  __shared__ int   sidx[256];
  __shared__ int   sel[10];
  const int myidx = idx[i];
  const float* row = WP + (size_t)i * n;
  for (int m = 0; m < 10; ++m) {
    float best = -1.0e30f; int bj = n;
    for (int j = tid; j < n; j += 256) {
      bool taken = false;
      for (int q = 0; q < m; ++q) taken |= (sel[q] == j);
      if (taken) continue;
      float v = (idx[j] == myidx) ? 1.0f : row[j];
      if (v > best || (v == best && j < bj)) { best = v; bj = j; }
    }
    sval[tid] = best; sidx[tid] = bj; __syncthreads();
    for (int st = 128; st > 0; st >>= 1) {
      if (tid < st) {
        if (sval[tid + st] > sval[tid] ||
            (sval[tid + st] == sval[tid] && sidx[tid + st] < sidx[tid])) {
          sval[tid] = sval[tid + st]; sidx[tid] = sidx[tid + st];
        }
      }
      __syncthreads();
    }
    if (tid == 0) {
      sel[m] = sidx[0];
      topk[i * 10 + m] = sidx[0];
      WNN[(size_t)i * n + sidx[0]] = 1;
    }
    __syncthreads();
  }
}

__global__ __launch_bounds__(256)
void k_v_deg(const unsigned char* __restrict__ WNN, unsigned char* __restrict__ V,
             float* __restrict__ deg, int n) {
  const int i = blockIdx.x;
  const int tid = threadIdx.x;
  __shared__ float red[256];
  int lsum = 0;
  for (int j = tid; j < n; j += 256) {
    unsigned char v = WNN[(size_t)i * n + j] & WNN[(size_t)j * n + i];
    V[(size_t)i * n + j] = v;
    lsum += v;
  }
  red[tid] = (float)lsum; __syncthreads();
  for (int st = 128; st > 0; st >>= 1) { if (tid < st) red[tid] += red[tid + st]; __syncthreads(); }
  if (tid == 0) deg[i] = red[0];
}

__global__ __launch_bounds__(256)
void k_gemm_u8_syrk(const unsigned char* __restrict__ V, int* __restrict__ VV, int n) {
  __shared__ __align__(16) unsigned char ldsA[128 * 64];
  __shared__ __align__(16) unsigned char ldsB[64 * 64];
  const int lane = threadIdx.x & 31;
  const int wave = threadIdx.x >> 5;
  const int h   = lane >> 4;
  const int l16 = lane & 15;
  const int wm = wave & 3, wn = wave >> 2;
  const int mrow0 = blockIdx.x * 128;
  const int nrow0 = blockIdx.y * 64;

#ifdef HAVE_TDM
  const unsigned ldsA_off = (unsigned)(size_t)&ldsA[0];
  const unsigned ldsB_off = (unsigned)(size_t)&ldsB[0];
#endif

  v8i acc[2][2] = {};
  for (int k = 0; k < n; k += 64) {
#ifdef HAVE_TDM
    if (threadIdx.x == 0) {
      tdm_load_tile_u8(V + (size_t)mrow0 * n + k, ldsA_off, 128u, (unsigned)n);
      tdm_load_tile_u8(V + (size_t)nrow0 * n + k, ldsB_off, 64u, (unsigned)n);
      __builtin_amdgcn_s_wait_tensorcnt(0);
    }
#else
    stage_tile_u8(V + (size_t)mrow0 * n + k, ldsA, 128, n, threadIdx.x);
    stage_tile_u8(V + (size_t)nrow0 * n + k, ldsB, 64, n, threadIdx.x);
#endif
    __syncthreads();

    v8i a[2], b[2];
#pragma unroll
    for (int i = 0; i < 2; ++i) {
      const unsigned char* ap = ldsA + (size_t)(wm * 32 + 16 * i + l16) * 64;
      int2 c0 = *(const int2*)(ap + h * 8);
      int2 c1 = *(const int2*)(ap + 16 + h * 8);
      int2 c2 = *(const int2*)(ap + 32 + h * 8);
      int2 c3 = *(const int2*)(ap + 48 + h * 8);
      a[i][0] = c0.x; a[i][1] = c0.y; a[i][2] = c1.x; a[i][3] = c1.y;
      a[i][4] = c2.x; a[i][5] = c2.y; a[i][6] = c3.x; a[i][7] = c3.y;
    }
#pragma unroll
    for (int j = 0; j < 2; ++j) {
      const unsigned char* bp = ldsB + (size_t)(wn * 32 + 16 * j + l16) * 64;
      int2 d0 = *(const int2*)(bp + h * 8);
      int2 d1 = *(const int2*)(bp + 16 + h * 8);
      int2 d2 = *(const int2*)(bp + 32 + h * 8);
      int2 d3 = *(const int2*)(bp + 48 + h * 8);
      b[j][0] = d0.x; b[j][1] = d0.y; b[j][2] = d1.x; b[j][3] = d1.y;
      b[j][4] = d2.x; b[j][5] = d2.y; b[j][6] = d3.x; b[j][7] = d3.y;
    }
#pragma unroll
    for (int i = 0; i < 2; ++i)
#pragma unroll
      for (int j = 0; j < 2; ++j)
        acc[i][j] = __builtin_amdgcn_wmma_i32_16x16x64_iu8(
            false, a[i], false, b[j], acc[i][j], false, false);

    __syncthreads();
  }

#pragma unroll
  for (int i = 0; i < 2; ++i)
#pragma unroll
    for (int j = 0; j < 2; ++j) {
      int col = nrow0 + wn * 32 + 16 * j + l16;
#pragma unroll
      for (int r = 0; r < 8; ++r) {
        int row = mrow0 + wm * 32 + 16 * i + h * 8 + r;
        VV[(size_t)row * n + col] = acc[i][j][r];
      }
    }
}

__global__ __launch_bounds__(256)
void k_wctilda(const unsigned char* __restrict__ V, const int* __restrict__ VVi,
               float* __restrict__ WCt, const float* __restrict__ deg,
               unsigned int n, unsigned int total) {
  for (unsigned int e = blockIdx.x * 256u + threadIdx.x; e < total; e += gridDim.x * 256u) {
    unsigned int i = e / n;
    float out = V[e] ? ((float)VVi[e] / fmaxf(deg[i], 1.0f)) : 0.0f;
    WCt[e] = out;
  }
}

__global__ __launch_bounds__(256)
void k_wchat(const float* __restrict__ WCt, const int* __restrict__ topk,
             float* __restrict__ WCh, int n) {
  const int i = blockIdx.y;
  const int j = blockIdx.x * 256 + threadIdx.x;
  __shared__ int t5[5];
  if (threadIdx.x < 5) t5[threadIdx.x] = topk[i * 10 + threadIdx.x];
  __syncthreads();
  float s = 0.f;
#pragma unroll
  for (int m = 0; m < 5; ++m) s += WCt[(size_t)t5[m] * n + j];
  WCh[(size_t)i * n + j] = s * 0.2f;
}

__global__ __launch_bounds__(256)
void k_loss(const float* __restrict__ Sd, const float* __restrict__ rmean,
            const float* __restrict__ WP, const float* __restrict__ WCh,
            float* __restrict__ partial, unsigned int n) {
  const unsigned int total = n * n;
  const unsigned int tid = threadIdx.x;
  __shared__ float red[256];
  float lsum = 0.f;
  for (unsigned int e = blockIdx.x * 256u + tid; e < total; e += gridDim.x * 256u) {
    unsigned int i = e / n;
    unsigned int j = e - i * n;
    if (i == j) continue;
    float sd = Sd[e] / rmean[i];
    float wc = 0.5f * (WCh[e] + WCh[(size_t)j * n + i]);
    float W  = 0.5f * (WP[e] + wc);
    float pull = sd * sd * W;
    float td = fmaxf(1.0f - sd, 0.0f);
    float push = td * td * (1.0f - W);
    lsum += pull + push;
  }
  red[tid] = lsum; __syncthreads();
  for (int st = 128; st > 0; st >>= 1) { if (tid < st) red[tid] += red[tid + st]; __syncthreads(); }
  if (tid == 0) partial[blockIdx.x] = red[0];
}

__global__ __launch_bounds__(256)
void k_loss_final(const float* __restrict__ partial, float* __restrict__ out,
                  int nb, float denom) {
  const int tid = threadIdx.x;
  __shared__ float red[256];
  float lsum = 0.f;
  for (int b = tid; b < nb; b += 256) lsum += partial[b];
  red[tid] = lsum; __syncthreads();
  for (int st = 128; st > 0; st >>= 1) { if (tid < st) red[tid] += red[tid + st]; __syncthreads(); }
  if (tid == 0) out[0] = red[0] / denom;
}

extern "C" void kernel_launch(void* const* d_in, const int* in_sizes, int n_in,
                              void* d_out, int out_size, void* d_ws, size_t ws_size,
                              hipStream_t stream) {
  const float* s_emb = (const float*)d_in[0];
  const float* t_emb = (const float*)d_in[1];
  const int*   idx   = (const int*)d_in[2];
  float* out = (float*)d_out;

  const int N = NROWS, D = NDIM;
  char* w = (char*)d_ws;
  const size_t MB = (size_t)1 << 20;

  unsigned short* sbf = (unsigned short*)(w + 0 * MB);
  unsigned short* tbf = (unsigned short*)(w + 4 * MB);
  float* sq      = (float*)(w + 8 * MB);
  float* rmean   = (float*)(w + 8 * MB + 64 * 1024);
  float* deg     = (float*)(w + 8 * MB + 128 * 1024);
  int*   topk    = (int*)  (w + 8 * MB + 192 * 1024);
  float* partial = (float*)(w + 8 * MB + 384 * 1024);
  float* Gs      = (float*)(w + 16 * MB);
  float* Gt      = (float*)(w + 80 * MB);
  unsigned char* WNN = (unsigned char*)(w + 144 * MB);
  unsigned char* V   = (unsigned char*)(w + 160 * MB);
  int*   VVi     = (int*)(w + 176 * MB);
  float* WCh     = (float*)(w + 240 * MB);

  const unsigned int total = (unsigned int)N * (unsigned int)N;

  k_prep<<<N, 256, 0, stream>>>(s_emb, t_emb, sbf, tbf, sq, N, D);

  dim3 ggrid(N / 128, N / 64);
  k_gemm_bf16_syrk<<<ggrid, 256, 0, stream>>>((const __bf16*)sbf, Gs, N, D);
  k_gemm_bf16_syrk<<<ggrid, 256, 0, stream>>>((const __bf16*)tbf, Gt, N, D);

  k_sdist<<<N, 256, 0, stream>>>(Gs, sq, rmean, N);
  k_wp<<<2048, 256, 0, stream>>>(Gt, total);

  hipMemsetAsync(WNN, 0, (size_t)N * N, stream);
  k_topk<<<N, 256, 0, stream>>>(Gt, idx, topk, WNN, N);
  k_v_deg<<<N, 256, 0, stream>>>(WNN, V, deg, N);

  k_gemm_u8_syrk<<<ggrid, 256, 0, stream>>>(V, VVi, N);
  k_wctilda<<<2048, 256, 0, stream>>>(V, VVi, (float*)VVi, deg, (unsigned int)N, total);

  dim3 hgrid(N / 256, N);
  k_wchat<<<hgrid, 256, 0, stream>>>((const float*)VVi, topk, WCh, N);

  k_loss<<<1024, 256, 0, stream>>>(Gs, rmean, Gt, WCh, partial, (unsigned int)N);
  k_loss_final<<<1, 256, 0, stream>>>(partial, out, 1024, (float)N * (float)(N - 1));
}
